// Attention_41884521071130
// MI455X (gfx1250) — matrix operand order repaired and checked by reading, not compiled, not run on hardware
//
#include <hip/hip_runtime.h>
#include <hip/hip_bf16.h>
#include <math.h>

typedef __attribute__((ext_vector_type(16))) _Float16 v16h;
typedef __attribute__((ext_vector_type(8)))  float    v8f;

#define NPTS   4096
#define CDIM   256
#define HEADS  8
#define HD     32
#define KNN    16
#define KT     (CDIM / 32)   // 8 K-steps of 32

// ---------------------------------------------------------------------------
// Fragment swizzle mappings (CDNA5 ISA 7.12.2, wave32, 16x16x32 f16):
//  A (16x32): lane = m + 16*g where g = K-subgroup; element e holds
//             K = 32*kk + (g?8:0) + (e<8 ? e : e+8).
//  B (32x16): lane = (n&15) + 16*half; element e holds K = 32*kk + (half?8:0) + (e<8 ? e : e+8).
// Stored so one lane's 16 f16 elements are 32 contiguous bytes
// (=> 2x ds_load_b128 / global_load_b128 per fragment, zero loop-side ALU).
// ---------------------------------------------------------------------------

// Pre-swizzle a fp32 weight matrix W[256 x Nout] into f16 B-fragment order:
// out[ ((nt*KT + kk)*32 + lane)*16 + e ]   (nt = 16-wide column tile index)
__global__ void __launch_bounds__(256)
swizzle_weights(const float* __restrict__ W, _Float16* __restrict__ out, int Nout) {
    const int d     = blockIdx.x * 256 + threadIdx.x;   // dst index
    const int e     = d & 15;
    const int lane  = (d >> 4) & 31;
    const int kkidx = d >> 9;
    const int kk    = kkidx & (KT - 1);
    const int nt    = kkidx >> 3;
    const int half  = lane >> 4;
    const int n     = nt * 16 + (lane & 15);
    const int k     = kk * 32 + (e < 8 ? half * 8 + e : 16 + half * 8 + (e - 8));
    out[d] = (_Float16)W[(size_t)k * Nout + n];
}

// ---- GEMM: Out[M x Nout] = A[M x 256] * W(preswizzled f16) + bias ---------
// block = 256 threads (8 waves); block tile = 32 rows x 256 cols.
// Each wave owns 2 row tiles x 2 col tiles -> 4 WMMAs per K-step while
// loading only 2 A frags (LDS) + 2 B frags (global). grid=(M/32, Nout/256)
__global__ void __launch_bounds__(256)
gemm_f16wmma(const float* __restrict__ A, const _Float16* __restrict__ Wsw,
             const float* __restrict__ bias, float* __restrict__ Out, int Nout) {
    __shared__ _Float16 Ash[32 * CDIM] __attribute__((aligned(32)));  // 2 swizzled tiles
    const int row0 = blockIdx.x * 32;
    const int tid  = threadIdx.x;

    // Stage + swizzle the 32x256 A tile (coalesced global reads, one col/thread).
    {
        const int k  = tid;
        const int kk = k >> 5, r = k & 31;
        const int g  = (r >> 3) & 1;
        const int e  = (r & 7) | ((r >> 4) << 3);
        const int base = (kk * 32 + 16 * g) * 16 + e;
#pragma unroll
        for (int m = 0; m < 16; ++m) {
            Ash[base + m * 16] =
                (_Float16)A[(size_t)(row0 + m) * CDIM + k];
            Ash[16 * CDIM + base + m * 16] =
                (_Float16)A[(size_t)(row0 + 16 + m) * CDIM + k];
        }
    }
    __syncthreads();

    const int wave = tid >> 5, lane = tid & 31;
    const int nt0  = blockIdx.y * 16 + wave * 2;   // global 16-col tile index

    v8f acc00 = {}, acc01 = {}, acc10 = {}, acc11 = {};
#pragma unroll
    for (int kk = 0; kk < KT; ++kk) {
        const v16h a0 = *(const v16h*)(Ash + (kk * 32 + lane) * 16);
        const v16h a1 = *(const v16h*)(Ash + 16 * CDIM + (kk * 32 + lane) * 16);
        const v16h b0 = *(const v16h*)(Wsw + ((size_t)(nt0 * KT + kk) * 32 + lane) * 16);
        const v16h b1 = *(const v16h*)(Wsw + ((size_t)((nt0 + 1) * KT + kk) * 32 + lane) * 16);
        acc00 = __builtin_amdgcn_wmma_f32_16x16x32_f16(false, a0, false, b0,
                                                       (short)0, acc00, false, false);
        acc01 = __builtin_amdgcn_wmma_f32_16x16x32_f16(false, a0, false, b1,
                                                       (short)0, acc01, false, false);
        acc10 = __builtin_amdgcn_wmma_f32_16x16x32_f16(false, a1, false, b0,
                                                       (short)0, acc10, false, false);
        acc11 = __builtin_amdgcn_wmma_f32_16x16x32_f16(false, a1, false, b1,
                                                       (short)0, acc11, false, false);
    }

    const int half = lane >> 4;
    const int ncA  = nt0 * 16 + (lane & 15);
    const int ncB  = ncA + 16;
    const float bvA = bias ? bias[ncA] : 0.0f;
    const float bvB = bias ? bias[ncB] : 0.0f;
#pragma unroll
    for (int r = 0; r < 8; ++r) {
        const int m0 = row0 + half * 8 + r;
        Out[(size_t)m0 * Nout + ncA]        = acc00[r] + bvA;
        Out[(size_t)m0 * Nout + ncB]        = acc01[r] + bvB;
        Out[(size_t)(m0 + 16) * Nout + ncA] = acc10[r] + bvA;
        Out[(size_t)(m0 + 16) * Nout + ncB] = acc11[r] + bvB;
    }
}

// ---- Fused gather + pos-MLP (WMMA) + attention, one block per point -------
__global__ void __launch_bounds__(256)
fused_point_attn(const float* __restrict__ xy, const float* __restrict__ qkv,
                 const int* __restrict__ nns,
                 const float* __restrict__ w1, const float* __restrict__ b1,
                 const _Float16* __restrict__ w2sw, const float* __restrict__ b2,
                 float* __restrict__ xpre, float* __restrict__ out_attn,
                 float* __restrict__ out_knn) {
    __shared__ int       nbr[KNN];
    __shared__ float     dlt[KNN][2];
    __shared__ _Float16  h1sw[16 * CDIM] __attribute__((aligned(32))); // A-frag order
    __shared__ float     kf[KNN][CDIM];
    __shared__ float     vpos[KNN][CDIM];
    __shared__ float     qv[CDIM];
    __shared__ float     prob[HEADS][KNN];

    const int bn  = blockIdx.x;            // b*N + n
    const int b   = bn >> 12;
    const int n   = bn & (NPTS - 1);
    const int tid = threadIdx.x;

    if (tid < KNN) {
        const int idx = nns[(size_t)bn * 32 + tid];   // first 16 of 32 neighbors
        nbr[tid] = idx;
        out_knn[(size_t)bn * KNN + tid] = (float)idx;
        const float x0 = xy[(size_t)bn * 2 + 0];
        const float y0 = xy[(size_t)bn * 2 + 1];
        dlt[tid][0] = x0 - xy[(size_t)(b * NPTS + idx) * 2 + 0];
        dlt[tid][1] = y0 - xy[(size_t)(b * NPTS + idx) * 2 + 1];
    }
    qv[tid] = qkv[(size_t)bn * (3 * CDIM) + tid];
    __syncthreads();

    // h1 = relu(delta@W1 + b1), written directly in swizzled A-fragment order.
    {
        const float w1a = w1[tid], w1b = w1[CDIM + tid], bb = b1[tid];
        const int k  = tid;
        const int kk = k >> 5, rr = k & 31;
        const int g  = (rr >> 3) & 1;
        const int e  = (rr & 7) | ((rr >> 4) << 3);
        const int base = (kk * 32 + 16 * g) * 16 + e;
#pragma unroll
        for (int r = 0; r < KNN; ++r) {
            const float h = dlt[r][0] * w1a + dlt[r][1] * w1b + bb;
            h1sw[base + r * 16] = (_Float16)fmaxf(h, 0.0f);
        }
    }
    // Gather k / v rows of the 16 neighbors (coalesced per row).
#pragma unroll
    for (int r = 0; r < KNN; ++r) {
        const size_t base = (size_t)(b * NPTS + nbr[r]) * (3 * CDIM);
        kf[r][tid]   = qkv[base + CDIM + tid];
        vpos[r][tid] = qkv[base + 2 * CDIM + tid] + b2[tid];
    }
    __syncthreads();

    // pos GEMM: 16x256 (h1,f16) @ 256x256 (w2) accumulated into vpos.
    const int wave = tid >> 5, lane = tid & 31;
    {
        const int nt0 = wave * 2;
        v8f acc0 = {}, acc1 = {};
#pragma unroll
        for (int kk = 0; kk < KT; ++kk) {
            const v16h a  = *(const v16h*)(h1sw + (kk * 32 + lane) * 16);
            const v16h b0 = *(const v16h*)(w2sw + ((size_t)(nt0 * KT + kk) * 32 + lane) * 16);
            const v16h b1 = *(const v16h*)(w2sw + ((size_t)((nt0 + 1) * KT + kk) * 32 + lane) * 16);
            acc0 = __builtin_amdgcn_wmma_f32_16x16x32_f16(false, a, false, b0,
                                                          (short)0, acc0, false, false);
            acc1 = __builtin_amdgcn_wmma_f32_16x16x32_f16(false, a, false, b1,
                                                          (short)0, acc1, false, false);
        }
        const int half = lane >> 4;
        const int ncA  = nt0 * 16 + (lane & 15);
#pragma unroll
        for (int r = 0; r < 8; ++r) {
            vpos[half * 8 + r][ncA]      += acc0[r];   // exclusive (m,n) per lane
            vpos[half * 8 + r][ncA + 16] += acc1[r];
        }
    }
    __syncthreads();

    // Attention logits + softmax over 16 neighbors: wave == head.
    const int h  = wave;
    const int kn = lane & 15;                  // both half-waves compute identically
    float logit;
    {
        const float* kr = &kf[kn][h * HD];
        const float* qh = &qv[h * HD];
        float s = 0.0f;
#pragma unroll
        for (int d = 0; d < HD; ++d) s += qh[d] * kr[d];
        logit = s * 0.17677669529663687f;      // 1/sqrt(32)
    }
    float mx = logit;
#pragma unroll
    for (int off = 8; off; off >>= 1) mx = fmaxf(mx, __shfl_xor(mx, off, 16));
    const float e = __expf(logit - mx);
    float sm = e;
#pragma unroll
    for (int off = 8; off; off >>= 1) sm += __shfl_xor(sm, off, 16);
    const float p = e / sm;
    if (lane < 16) {
        prob[h][lane] = p;
        out_attn[(((size_t)(b * HEADS + h)) * NPTS + n) * KNN + lane] = p;
    }
    __syncthreads();

    // x_pre[c] = sum_k prob[head(c)][k] * vpos[k][c]
    {
        const int hh = tid >> 5;
        float s = 0.0f;
#pragma unroll
        for (int k = 0; k < KNN; ++k) s += prob[hh][k] * vpos[k][tid];
        xpre[(size_t)bn * CDIM + tid] = s;
    }
}

extern "C" void kernel_launch(void* const* d_in, const int* in_sizes, int n_in,
                              void* d_out, int out_size, void* d_ws, size_t ws_size,
                              hipStream_t stream) {
    const float* xy       = (const float*)d_in[0];
    const float* xy_embed = (const float*)d_in[1];
    const int*   nns      = (const int*)  d_in[2];
    const float* w_delta1 = (const float*)d_in[3];
    const float* b_delta1 = (const float*)d_in[4];
    const float* w_delta2 = (const float*)d_in[5];
    const float* b_delta2 = (const float*)d_in[6];
    const float* w_qkv    = (const float*)d_in[7];
    const float* w_proj   = (const float*)d_in[8];
    const float* b_proj   = (const float*)d_in[9];

    const int B = 4, M = B * NPTS;                 // 16384 rows

    // workspace layout (all 32B-aligned offsets):
    //   qkv f32 (16384x768) | xpre f32 (16384x256) | swizzled f16 weights
    char* ws = (char*)d_ws;
    float*     qkv_ws   = (float*)ws;
    float*     xpre     = (float*)(ws + (size_t)M * 768 * 4);
    _Float16*  wqkv_sw  = (_Float16*)(ws + (size_t)M * 768 * 4 + (size_t)M * CDIM * 4);
    _Float16*  wproj_sw = wqkv_sw + (size_t)CDIM * 768;
    _Float16*  wd2_sw   = wproj_sw + (size_t)CDIM * CDIM;

    // d_out: x (4,4096,256) | attn (4,8,4096,16) | knn_idx (4,4096,16)
    float* out_x    = (float*)d_out;
    float* out_attn = out_x + (size_t)M * CDIM;
    float* out_knn  = out_attn + (size_t)B * HEADS * NPTS * KNN;

    // 0) one-time weight conversion + fragment swizzle (f32 -> f16)
    swizzle_weights<<<768, 256, 0, stream>>>(w_qkv,    wqkv_sw, 768);
    swizzle_weights<<<256, 256, 0, stream>>>(w_proj,   wproj_sw, 256);
    swizzle_weights<<<256, 256, 0, stream>>>(w_delta2, wd2_sw,   256);

    // 1) qkv = xy_embed @ w_qkv            (M=16384, K=256, N=768)
    gemm_f16wmma<<<dim3(M / 32, 768 / 256), 256, 0, stream>>>(
        xy_embed, wqkv_sw, nullptr, qkv_ws, 768);

    // 2) fused gather + pos MLP + attention (one block per point)
    fused_point_attn<<<M, 256, 0, stream>>>(
        xy, qkv_ws, nns, w_delta1, b_delta1, wd2_sw, b_delta2,
        xpre, out_attn, out_knn);

    // 3) x = x_pre @ w_proj + b_proj       (M=16384, K=256, N=256)
    gemm_f16wmma<<<dim3(M / 32, 256 / 256), 256, 0, stream>>>(
        xpre, wproj_sw, b_proj, out_x, 256);
}
